// TransformerLayer_33913061769959
// MI455X (gfx1250) — matrix operand order repaired and checked by reading, not compiled, not run on hardware
//
#include <hip/hip_runtime.h>
#include <hip/hip_bf16.h>

// ---------------- problem constants ----------------
#define B_   4
#define N_   2048
#define D_   2048
#define H_   16
#define DK_  128
#define DFF_ 8192
#define EPS_ 1e-5f
static constexpr size_t SND = (size_t)B_ * N_ * D_;   // 16,777,216

typedef __attribute__((ext_vector_type(16))) _Float16 v16h;
typedef __attribute__((ext_vector_type(8)))  float    v8f;
typedef __attribute__((ext_vector_type(4)))  unsigned int v4u;
typedef __attribute__((ext_vector_type(8)))  int      v8i;
typedef __attribute__((ext_vector_type(4)))  int      v4i;

#if defined(__has_builtin)
#if __has_builtin(__builtin_amdgcn_tensor_load_to_lds) && __has_builtin(__builtin_amdgcn_s_wait_tensorcnt)
#define USE_TDM 1
#endif
#endif
#ifndef USE_TDM
#define USE_TDM 0
#endif

// D = A x B + C  (16x16x32, f16 in, f32 accum)
__device__ __forceinline__ v8f wmma16(v16h a, v16h b, v8f c) {
    return __builtin_amdgcn_wmma_f32_16x16x32_f16(false, a, false, b, (short)0, c, false, false);
}

// Build a v16h fragment from two 16-byte LDS chunks (elements 0..7 and 8..15).
__device__ __forceinline__ v16h load_frag2(const _Float16* p0, const _Float16* p1) {
    union { v16h v; uint4 u[2]; } t;
    t.u[0] = *(const uint4*)p0;
    t.u[1] = *(const uint4*)p1;
    return t.v;
}

#if USE_TDM
// Tensor Data Mover: 2D tile of f16 (tile_w elems per row, tile_h rows, row
// stride stride_elems) from global -> LDS, contiguous rows in LDS.
// D# packing per CDNA5 ISA section 8.3/8.4 (group0: count/lds/global/type,
// group1: data_size + dims + dim0 stride; 2D tile so groups 2/3 are zero).
__device__ __forceinline__ void tdm_load_2d(const void* gptr, void* lptr,
                                            unsigned tile_w, unsigned tile_h,
                                            unsigned stride_elems) {
    unsigned long long ga = (unsigned long long)(uintptr_t)gptr;
    v4u g0;
    g0[0] = 1u;                                              // count=1, user mode
    g0[1] = (unsigned)(uintptr_t)lptr;                       // lds_addr (bytes)
    g0[2] = (unsigned)ga;                                    // global_addr[31:0]
    g0[3] = ((unsigned)(ga >> 32) & 0x01FFFFFFu) | (2u << 30); // addr[56:32] | type=2
    v8i g1;
    g1[0] = (int)(1u << 16);                                 // data_size=1 -> 2 bytes
    g1[1] = (int)((tile_w & 0xFFFFu) << 16);                 // tensor_dim0[15:0]
    g1[2] = (int)((tile_w >> 16) | ((tile_h & 0xFFFFu) << 16)); // td0 hi | td1 lo
    g1[3] = (int)((tile_h >> 16) | (tile_w << 16));          // td1 hi | tile_dim0
    g1[4] = (int)tile_h;                                     // tile_dim1 (tile_dim2=0)
    g1[5] = (int)stride_elems;                               // tensor_dim0_stride lo
    g1[6] = 0;
    g1[7] = 0;
    v4i z = {0, 0, 0, 0};
#if __clang_major__ >= 23
    v8i z8 = {0, 0, 0, 0, 0, 0, 0, 0};
    __builtin_amdgcn_tensor_load_to_lds(g0, g1, z, z, z8, 0);
#else
    __builtin_amdgcn_tensor_load_to_lds(g0, g1, z, z, 0);
#endif
}
#endif

// ---------------- weight convert + transpose: fp32 W[K][N] -> f16 Wt[N][K] ----------------
__global__ __launch_bounds__(256)
void wcvt_kernel(const float* __restrict__ W, _Float16* __restrict__ Wt, int K, int N) {
    __shared__ _Float16 t[32][33];
    const int k0 = blockIdx.x * 32, n0 = blockIdx.y * 32;
    const int tid = threadIdx.x;
#pragma unroll
    for (int i = 0; i < 4; ++i) {
        const int id = i * 256 + tid;
        const int kr = id >> 5, nc = id & 31;
        t[nc][kr] = (_Float16)W[(size_t)(k0 + kr) * N + n0 + nc];
    }
    __syncthreads();
#pragma unroll
    for (int i = 0; i < 4; ++i) {
        const int id = i * 256 + tid;
        const int nr = id >> 5, kc = id & 31;
        Wt[(size_t)(n0 + nr) * K + k0 + kc] = t[nr][kc];
    }
}

// ---------------- LayerNorm (ddof=1, eps added to std) : fp32 -> f16 ----------------
__global__ __launch_bounds__(256)
void ln_kernel(const float* __restrict__ x, const float* __restrict__ scale,
               const float* __restrict__ shift, _Float16* __restrict__ out) {
    const int row = blockIdx.x;
    const float* xr = x + (size_t)row * D_;
    float s = 0.f, s2 = 0.f;
    for (int i = threadIdx.x; i < D_; i += 256) { float v = xr[i]; s += v; s2 += v * v; }
#pragma unroll
    for (int off = 16; off > 0; off >>= 1) {
        s  += __shfl_xor(s,  off, 32);
        s2 += __shfl_xor(s2, off, 32);
    }
    __shared__ float red[16];
    const int wid = threadIdx.x >> 5;
    if ((threadIdx.x & 31) == 0) { red[wid * 2] = s; red[wid * 2 + 1] = s2; }
    __syncthreads();
    float ts = 0.f, ts2 = 0.f;
#pragma unroll
    for (int w = 0; w < 8; ++w) { ts += red[w * 2]; ts2 += red[w * 2 + 1]; }
    const float mu  = ts / (float)D_;
    const float var = fmaxf(0.f, (ts2 - (float)D_ * mu * mu) / (float)(D_ - 1));
    const float inv = 1.f / (sqrtf(var) + EPS_);
    for (int i = threadIdx.x; i < D_; i += 256)
        out[(size_t)row * D_ + i] = (_Float16)(((xr[i] - mu) * inv) * scale[i] + shift[i]);
}

// ---------------- WMMA GEMM:  C[M,N] = epilogue(A_f16[M,K] @ Bt_f16[N][K]^T) ----------------
// Block tile 128x128, K-step 32, double-buffered LDS (TDM prefetch overlaps WMMA).
// 8 waves: wave (wm 0..3, wn 0..1) owns 32x64 (2x4 WMMA tiles).
template<bool BIAS, bool RES, bool GELU_, typename OutT>
__global__ __launch_bounds__(256)
void gemm_kernel(const _Float16* __restrict__ A, const _Float16* __restrict__ Bt,
                 const float* __restrict__ bias, const float* __restrict__ res,
                 OutT* __restrict__ C, int M, int K, int N) {
    __shared__ _Float16 sA[2][128 * 32];   // [m][k]
    __shared__ _Float16 sB[2][128 * 32];   // [n][k]
    const int tid = threadIdx.x, lane = tid & 31, wid = tid >> 5;
    const int wm = wid & 3, wn = wid >> 2;
    const int l16 = lane & 15, khalf = lane >> 4;
    const int m0 = blockIdx.x * 128, n0 = blockIdx.y * 128;
    const int nk = K / 32;
    (void)M;

    v8f acc[2][4] = {};

    auto stage = [&](int buf, int kblk) {
        const int k0 = kblk * 32;
#if USE_TDM
        if (wid == 0) {
            tdm_load_2d(A  + (size_t)m0 * K + k0, sA[buf], 32, 128, (unsigned)K);
            tdm_load_2d(Bt + (size_t)n0 * K + k0, sB[buf], 32, 128, (unsigned)K);
        }
#else
        const int row = tid >> 1, half = tid & 1;
        {
            const _Float16* src = A + (size_t)(m0 + row) * K + k0 + half * 16;
            uint4* dst = (uint4*)(sA[buf] + row * 32 + half * 16);
            dst[0] = ((const uint4*)src)[0];
            dst[1] = ((const uint4*)src)[1];
        }
        {
            const _Float16* src = Bt + (size_t)(n0 + row) * K + k0 + half * 16;
            uint4* dst = (uint4*)(sB[buf] + row * 32 + half * 16);
            dst[0] = ((const uint4*)src)[0];
            dst[1] = ((const uint4*)src)[1];
        }
#endif
    };

    stage(0, 0);
    for (int i = 0; i < nk; ++i) {
        const int buf = i & 1;
#if USE_TDM
        if (wid == 0) __builtin_amdgcn_s_wait_tensorcnt(0);  // tiles(i) complete
#endif
        __syncthreads();                                      // publish buf; prior readers of buf^1 done
        if (i + 1 < nk) stage(buf ^ 1, i + 1);                // prefetch overlaps compute

        v16h af[2];
#pragma unroll
        for (int ms = 0; ms < 2; ++ms) {
            const int row = wm * 32 + ms * 16 + l16;
            af[ms] = load_frag2(sA[buf] + row * 32 + khalf * 8,
                                sA[buf] + row * 32 + 16 + khalf * 8);
        }
#pragma unroll
        for (int ns = 0; ns < 4; ++ns) {
            const int n = wn * 64 + ns * 16 + l16;
            v16h bf = load_frag2(sB[buf] + n * 32 + khalf * 8,
                                 sB[buf] + n * 32 + 16 + khalf * 8);
#pragma unroll
            for (int ms = 0; ms < 2; ++ms)
                acc[ms][ns] = wmma16(af[ms], bf, acc[ms][ns]);
        }
    }

    // epilogue: lane holds col n = l16, rows r + 8*khalf of each 16x16 tile
#pragma unroll
    for (int ms = 0; ms < 2; ++ms) {
#pragma unroll
        for (int ns = 0; ns < 4; ++ns) {
            const int ng = n0 + wn * 64 + ns * 16 + l16;
            const float bv = BIAS ? bias[ng] : 0.f;
            const int mbase = m0 + wm * 32 + ms * 16 + khalf * 8;
#pragma unroll
            for (int r = 0; r < 8; ++r) {
                const int mg = mbase + r;
                float v = acc[ms][ns][r] + bv;
                if (GELU_) v = 0.5f * v * (1.f + erff(v * 0.70710678118654752f));
                if (RES)   v += res[(size_t)mg * N + ng];
                C[(size_t)mg * N + ng] = (OutT)v;
            }
        }
    }
}

// ---------------- Flash attention (causal + pad mask, online softmax) ----------------
// Grid: (N/64, H, B). Block: 128 threads = 4 waves; wave owns 16 q-rows of a 64-row block.
#define VTP 72   // padded key-stride of transposed V tile (keeps 16B alignment, cuts conflicts)
__global__ __launch_bounds__(128)
void attn_kernel(const _Float16* __restrict__ qm, const _Float16* __restrict__ km,
                 const _Float16* __restrict__ vm, const float* __restrict__ amask,
                 _Float16* __restrict__ ctx) {
    const int bq = blockIdx.x;           // q block (64 rows)
    const int h  = blockIdx.y;
    const int b  = blockIdx.z;
    const int tid = threadIdx.x, lane = tid & 31, wid = tid >> 5;
    const int l16 = lane & 15, khalf = lane >> 4;
    const float SCL = 0.0883883476483184f;   // 1/sqrt(128)
    const float NEG = -__builtin_inff();

    __shared__ _Float16 Qs[64 * 128];        // [qr][dim]
    __shared__ _Float16 Ks[64 * 128];        // [key][dim]
    __shared__ _Float16 Vts[128 * VTP];      // [dim][key] (transposed, padded)
    __shared__ _Float16 Ps[4][16 * 64];      // per-wave P tile [row][key]

    const size_t qoff = ((size_t)b * N_ + (size_t)bq * 64) * D_ + (size_t)h * DK_;
#if USE_TDM
    if (wid == 0) tdm_load_2d(qm + qoff, Qs, 128, 64, (unsigned)D_);
#else
#pragma unroll
    for (int c = 0; c < 8; ++c) {            // 64x128 f16
        const int id = c * 128 + tid;
        const int r = id >> 4, ck = id & 15;
        *(uint4*)(Qs + r * 128 + ck * 8) =
            *(const uint4*)(qm + qoff + (size_t)r * D_ + ck * 8);
    }
#endif

    float m_run[8], l_run[8];
    v8f accO[8];
#pragma unroll
    for (int r = 0; r < 8; ++r) { m_run[r] = NEG; l_run[r] = 0.f; }
#pragma unroll
    for (int dt = 0; dt < 8; ++dt) accO[dt] = (v8f){0,0,0,0,0,0,0,0};

    for (int kt = 0; kt <= bq; ++kt) {
        const size_t koff = ((size_t)b * N_ + (size_t)kt * 64) * D_ + (size_t)h * DK_;
#if USE_TDM
        if (wid == 0) tdm_load_2d(km + koff, Ks, 128, 64, (unsigned)D_);
#else
#pragma unroll
        for (int c = 0; c < 8; ++c) {        // K tile 64x128
            const int id = c * 128 + tid;
            const int r = id >> 4, ck = id & 15;
            *(uint4*)(Ks + r * 128 + ck * 8) =
                *(const uint4*)(km + koff + (size_t)r * D_ + ck * 8);
        }
#endif
#pragma unroll
        for (int c = 0; c < 8; ++c) {        // V tile transposed into Vts[dim][key]
            const int id = c * 128 + tid;    // 1024 chunks of 8 halves
            const int key = id >> 4, ck = id & 15;
            union { uint4 u; _Float16 hx[8]; } tv;
            tv.u = *(const uint4*)(vm + koff + (size_t)key * D_ + ck * 8);
#pragma unroll
            for (int e = 0; e < 8; ++e)
                Vts[(ck * 8 + e) * VTP + key] = tv.hx[e];
        }
#if USE_TDM
        if (wid == 0) __builtin_amdgcn_s_wait_tensorcnt(0);
#endif
        __syncthreads();

        // ---- S = Q x K^T  (16 q-rows x 64 keys) ----
        v8f sc[4] = {};
#pragma unroll
        for (int kk = 0; kk < 4; ++kk) {
            const int qrow = wid * 16 + l16;
            v16h aq = load_frag2(Qs + qrow * 128 + kk * 32 + khalf * 8,
                                 Qs + qrow * 128 + kk * 32 + 16 + khalf * 8);
#pragma unroll
            for (int ns = 0; ns < 4; ++ns) {
                const int krow = ns * 16 + l16;
                v16h bk = load_frag2(Ks + krow * 128 + kk * 32 + khalf * 8,
                                     Ks + krow * 128 + kk * 32 + 16 + khalf * 8);
                sc[ns] = wmma16(aq, bk, sc[ns]);
            }
        }

        bool kvalid[4];
#pragma unroll
        for (int ns = 0; ns < 4; ++ns)
            kvalid[ns] = amask[(size_t)b * N_ + kt * 64 + ns * 16 + l16] != 0.f;

        // ---- masked online softmax (row spread across 16 lanes) ----
#pragma unroll
        for (int r = 0; r < 8; ++r) {
            const int qg = bq * 64 + wid * 16 + khalf * 8 + r;
            float mx = NEG;
#pragma unroll
            for (int ns = 0; ns < 4; ++ns) {
                const int kg = kt * 64 + ns * 16 + l16;
                float sv = sc[ns][r] * SCL;
                sv = (kvalid[ns] && kg <= qg) ? sv : NEG;
                sc[ns][r] = sv;
                mx = fmaxf(mx, sv);
            }
#pragma unroll
            for (int off = 1; off < 16; off <<= 1) mx = fmaxf(mx, __shfl_xor(mx, off, 32));
            const float mn   = fmaxf(m_run[r], mx);
            const float mexp = (mn == NEG) ? 0.f : mn;
            const float corr = __expf(m_run[r] - mexp);   // -> 0 when m_run == -inf
            float psum = 0.f;
#pragma unroll
            for (int ns = 0; ns < 4; ++ns) {
                const float p = __expf(sc[ns][r] - mexp);
                psum += p;
                Ps[wid][(khalf * 8 + r) * 64 + ns * 16 + l16] = (_Float16)p;
            }
#pragma unroll
            for (int off = 1; off < 16; off <<= 1) psum += __shfl_xor(psum, off, 32);
            l_run[r] = l_run[r] * corr + psum;
            m_run[r] = mn;
#pragma unroll
            for (int dt = 0; dt < 8; ++dt) accO[dt][r] = accO[dt][r] * corr;
        }

        // ---- O += P x V  (contraction over 64 keys, 2 K-steps) ----
#pragma unroll
        for (int kk = 0; kk < 2; ++kk) {
            v16h ap = load_frag2(&Ps[wid][l16 * 64 + kk * 32 + khalf * 8],
                                 &Ps[wid][l16 * 64 + kk * 32 + 16 + khalf * 8]);
#pragma unroll
            for (int dt = 0; dt < 8; ++dt) {
                const int dim = dt * 16 + l16;
                v16h bv = load_frag2(Vts + dim * VTP + kk * 32 + khalf * 8,
                                     Vts + dim * VTP + kk * 32 + 16 + khalf * 8);
                accO[dt] = wmma16(ap, bv, accO[dt]);
            }
        }
        __syncthreads();
    }

    // ---- normalize + store ctx (f16, [B,N,D] with head slice) ----
    const size_t obase = ((size_t)b * N_ + (size_t)bq * 64 + wid * 16) * D_ + (size_t)h * DK_;
#pragma unroll
    for (int r = 0; r < 8; ++r) {
        const float inv = 1.f / l_run[r];   // causal diagonal guarantees l_run > 0
        const int mrow = khalf * 8 + r;
#pragma unroll
        for (int dt = 0; dt < 8; ++dt)
            ctx[obase + (size_t)mrow * D_ + dt * 16 + l16] = (_Float16)(accO[dt][r] * inv);
    }
}

// ---------------- host launcher ----------------
extern "C" void kernel_launch(void* const* d_in, const int* in_sizes, int n_in,
                              void* d_out, int out_size, void* d_ws, size_t ws_size,
                              hipStream_t stream) {
    (void)in_sizes; (void)n_in; (void)out_size; (void)ws_size;
    const float* x     = (const float*)d_in[0];
    const float* amask = (const float*)d_in[1];
    const float* Wq    = (const float*)d_in[2];
    const float* Wk    = (const float*)d_in[3];
    const float* Wv    = (const float*)d_in[4];
    const float* Wo    = (const float*)d_in[5];
    const float* bo    = (const float*)d_in[6];
    const float* ln1s  = (const float*)d_in[7];
    const float* ln1b  = (const float*)d_in[8];
    const float* ln2s  = (const float*)d_in[9];
    const float* ln2b  = (const float*)d_in[10];
    const float* W1    = (const float*)d_in[11];
    const float* b1    = (const float*)d_in[12];
    const float* W2    = (const float*)d_in[13];
    const float* b2    = (const float*)d_in[14];
    float* out = (float*)d_out;

    // ws layout (bytes):
    // [0,2S)   xln f16      [2S,10S)  q|k|v|ctx f16 (later h1)   [10S,14S) x1 f32
    // [14S..)  transposed f16 weights: wq,wk,wv,wo (S/2 each), w1 (2S), w2 (2S)
    char* ws = (char*)d_ws;
    _Float16* xln = (_Float16*)ws;
    _Float16* big = (_Float16*)(ws + 2 * SND);
    _Float16* qh  = big;
    _Float16* kh  = big + SND;
    _Float16* vh  = big + 2 * SND;
    _Float16* ch  = big + 3 * SND;
    _Float16* h1  = big;                                    // reuse (same 8*SND bytes)
    float*    x1  = (float*)(ws + 10 * SND);
    _Float16* wqT = (_Float16*)(ws + 14 * SND);
    _Float16* wkT = (_Float16*)(ws + 14 * SND + SND / 2);
    _Float16* wvT = (_Float16*)(ws + 15 * SND);
    _Float16* woT = (_Float16*)(ws + 15 * SND + SND / 2);
    _Float16* w1T = (_Float16*)(ws + 16 * SND);
    _Float16* w2T = (_Float16*)(ws + 18 * SND);

    const int M = B_ * N_;                                  // 8192 tokens
    dim3 blk(256);
    dim3 gD(M / 128, D_ / 128);                             // N = 2048 GEMMs
    dim3 gF(M / 128, DFF_ / 128);                           // N = 8192 GEMM

    // one-shot weight convert + transpose (bandwidth-bound, hits L2 afterwards)
    wcvt_kernel<<<dim3(D_ / 32, D_ / 32), blk, 0, stream>>>(Wq, wqT, D_, D_);
    wcvt_kernel<<<dim3(D_ / 32, D_ / 32), blk, 0, stream>>>(Wk, wkT, D_, D_);
    wcvt_kernel<<<dim3(D_ / 32, D_ / 32), blk, 0, stream>>>(Wv, wvT, D_, D_);
    wcvt_kernel<<<dim3(D_ / 32, D_ / 32), blk, 0, stream>>>(Wo, woT, D_, D_);
    wcvt_kernel<<<dim3(D_ / 32, DFF_ / 32), blk, 0, stream>>>(W1, w1T, D_, DFF_);
    wcvt_kernel<<<dim3(DFF_ / 32, D_ / 32), blk, 0, stream>>>(W2, w2T, DFF_, D_);

    // x1 = x + MHA(LN1(x))
    ln_kernel<<<M, blk, 0, stream>>>(x, ln1s, ln1b, xln);
    gemm_kernel<false, false, false, _Float16><<<gD, blk, 0, stream>>>(xln, wqT, nullptr, nullptr, qh, M, D_, D_);
    gemm_kernel<false, false, false, _Float16><<<gD, blk, 0, stream>>>(xln, wkT, nullptr, nullptr, kh, M, D_, D_);
    gemm_kernel<false, false, false, _Float16><<<gD, blk, 0, stream>>>(xln, wvT, nullptr, nullptr, vh, M, D_, D_);
    attn_kernel<<<dim3(N_ / 64, H_, B_), dim3(128), 0, stream>>>(qh, kh, vh, amask, ch);
    gemm_kernel<true, true, false, float><<<gD, blk, 0, stream>>>(ch, woT, bo, x, x1, M, D_, D_);

    // out = x1 + W2 @ gelu(W1 @ LN2(x1) + b1) + b2
    ln_kernel<<<M, blk, 0, stream>>>(x1, ln2s, ln2b, xln);
    gemm_kernel<true, false, true, _Float16><<<gF, blk, 0, stream>>>(xln, w1T, b1, nullptr, h1, M, D_, DFF_);
    gemm_kernel<true, true, false, float><<<gD, blk, 0, stream>>>(h1, w2T, b2, x1, out, M, DFF_, D_);
}
